// FPSTokenizer_5403068858479
// MI455X (gfx1250) — matrix operand order repaired and checked by reading, not compiled, not run on hardware
//
#include <hip/hip_runtime.h>
#include <hip/hip_bf16.h>

#define NPTS 8192
#define NB   8
#define NT   128
#define NK   16
#define TOKD 768
#define GROWS (NB*NT*NK)   // 16384 gathered rows
#define FLT_BIG 3.402823466e38f

typedef __attribute__((ext_vector_type(16))) __bf16 v16bf;
typedef __attribute__((ext_vector_type(8)))  float  v8f;
typedef __attribute__((ext_vector_type(4)))  unsigned int u32x4;

union ABFrag { v16bf v; u32x4 q[2]; };

__device__ inline __bf16 f2bf(float f) {
  unsigned int u = __float_as_uint(f);
  u = u + 0x7FFFu + ((u >> 16) & 1u);          // round-to-nearest-even
  unsigned short h = (unsigned short)(u >> 16);
  union { unsigned short s; __bf16 b; } cvt; cvt.s = h; return cvt.b;
}
__device__ inline float bf2f(__bf16 b) {
  union { __bf16 b; unsigned short s; } cvt; cvt.b = b;
  return __uint_as_float(((unsigned int)cvt.s) << 16);
}

// ---------------------------------------------------------------------------
// 1) Farthest-point sampling: one workgroup per batch, 1024 threads (32 waves),
//    8 pts/thread. Arg-max via wave32 shuffle reduce + one 32-entry LDS stage.
// ---------------------------------------------------------------------------
__global__ __launch_bounds__(1024) void fps_kernel(
    const float* __restrict__ coords, const float* __restrict__ times,
    int* __restrict__ sel, float* __restrict__ cent, float* __restrict__ mask)
{
  __shared__ float s_v[32];
  __shared__ int   s_i[32];
  __shared__ float s_c[4];
  const int b = blockIdx.x, tid = threadIdx.x;
  const int lane = tid & 31, wid = tid >> 5;

  float px[8], py[8], pz[8], pw[8], md[8];
  int   pidx[8];
  #pragma unroll
  for (int j = 0; j < 8; ++j) {
    int p = tid + j * 1024;
    pidx[j] = p;
    size_t row = (size_t)b * NPTS + p;
    px[j] = coords[row*3+0]; py[j] = coords[row*3+1]; pz[j] = coords[row*3+2];
    pw[j] = times[row];
    md[j] = FLT_BIG;
  }

  // block sum: wave shuffle reduce -> 32 partials -> wave0 reduce
  auto redsum = [&](float v)->float {
    #pragma unroll
    for (int m = 16; m; m >>= 1) v += __shfl_xor(v, m, 32);
    if (lane == 0) s_v[wid] = v;
    __syncthreads();
    if (wid == 0) {
      float v2 = s_v[lane];
      #pragma unroll
      for (int m = 16; m; m >>= 1) v2 += __shfl_xor(v2, m, 32);
      if (lane == 0) s_v[0] = v2;
    }
    __syncthreads();
    float r = s_v[0];
    __syncthreads();
    return r;
  };
  // block argmax (ties -> lowest index)
  auto argmax = [&](float v, int i)->int {
    #pragma unroll
    for (int m = 16; m; m >>= 1) {
      float ov = __shfl_xor(v, m, 32); int oi = __shfl_xor(i, m, 32);
      if (ov > v || (ov == v && oi < i)) { v = ov; i = oi; }
    }
    if (lane == 0) { s_v[wid] = v; s_i[wid] = i; }
    __syncthreads();
    if (wid == 0) {
      float v2 = s_v[lane]; int i2 = s_i[lane];
      #pragma unroll
      for (int m = 16; m; m >>= 1) {
        float ov = __shfl_xor(v2, m, 32); int oi = __shfl_xor(i2, m, 32);
        if (ov > v2 || (ov == v2 && oi < i2)) { v2 = ov; i2 = oi; }
      }
      if (lane == 0) s_i[0] = i2;
    }
    __syncthreads();
    int r = s_i[0];
    __syncthreads();
    return r;
  };

  float sx=0,sy=0,sz=0,sw=0;
  #pragma unroll
  for (int j = 0; j < 8; ++j) { sx+=px[j]; sy+=py[j]; sz+=pz[j]; sw+=pw[j]; }
  const float inv = 1.0f / (float)NPTS;
  float mx = redsum(sx)*inv, my = redsum(sy)*inv, mz = redsum(sz)*inv, mw = redsum(sw)*inv;

  float bv = -FLT_BIG; int bi = NPTS;
  #pragma unroll
  for (int j = 0; j < 8; ++j) {
    float dx=px[j]-mx, dy=py[j]-my, dz=pz[j]-mz, dw=pw[j]-mw;
    float d2 = dx*dx+dy*dy+dz*dz+dw*dw;
    if (d2 > bv || (d2 == bv && pidx[j] < bi)) { bv = d2; bi = pidx[j]; }
  }
  int s = argmax(bv, bi);

  for (int t = 0; t < NT; ++t) {
    if (tid == 0) {
      size_t row = (size_t)b * NPTS + s;
      s_c[0]=coords[row*3]; s_c[1]=coords[row*3+1]; s_c[2]=coords[row*3+2]; s_c[3]=times[row];
    }
    __syncthreads();
    float cx=s_c[0], cy=s_c[1], cz=s_c[2], cw=s_c[3];
    #pragma unroll
    for (int j = 0; j < 8; ++j) {
      float dx=px[j]-cx, dy=py[j]-cy, dz=pz[j]-cz, dw=pw[j]-cw;
      float d2 = dx*dx+dy*dy+dz*dz+dw*dw;
      md[j] = fminf(md[j], d2);
      if (pidx[j] == s) md[j] = -FLT_BIG;
    }
    if (tid == 0) {
      sel[b*NT + t] = s;
      size_t o = ((size_t)b*NT + t) * 4;
      cent[o]=cx; cent[o+1]=cy; cent[o+2]=cz; cent[o+3]=cw;
    }
    __syncthreads();                 // s_c stable until here
    if (t == NT-1) break;
    float lv = -FLT_BIG; int li = NPTS;
    #pragma unroll
    for (int j = 0; j < 8; ++j)
      if (md[j] > lv || (md[j] == lv && pidx[j] < li)) { lv = md[j]; li = pidx[j]; }
    s = argmax(lv, li);
  }
  if (tid < NT) mask[b*NT + tid] = 1.0f;
}

// ---------------------------------------------------------------------------
// 2) KNN: one workgroup (8 waves) per centroid; 16 argmin passes over LDS
// ---------------------------------------------------------------------------
__global__ __launch_bounds__(256) void knn_kernel(
    const float* __restrict__ coords, const float* __restrict__ times,
    const float* __restrict__ cent, int* __restrict__ knn)
{
  __shared__ float d2s[NPTS];
  __shared__ float s_v[8];
  __shared__ int   s_i[8];
  __shared__ int   s_win;
  const int g = blockIdx.x, b = g >> 7, tid = threadIdx.x;
  const int lane = tid & 31, wid = tid >> 5;
  float cx=cent[g*4], cy=cent[g*4+1], cz=cent[g*4+2], cw=cent[g*4+3];

  for (int i = tid; i < NPTS; i += 256) {
    size_t row = (size_t)b * NPTS + i;
    float dx=coords[row*3]-cx, dy=coords[row*3+1]-cy, dz=coords[row*3+2]-cz, dw=times[row]-cw;
    d2s[i] = dx*dx+dy*dy+dz*dz+dw*dw;
  }
  __syncthreads();

  for (int kk = 0; kk < NK; ++kk) {
    float bv = FLT_BIG; int bi = NPTS;
    for (int i = tid; i < NPTS; i += 256) {
      float v = d2s[i];
      if (v < bv || (v == bv && i < bi)) { bv = v; bi = i; }
    }
    #pragma unroll
    for (int m = 16; m; m >>= 1) {
      float ov = __shfl_xor(bv, m, 32); int oi = __shfl_xor(bi, m, 32);
      if (ov < bv || (ov == bv && oi < bi)) { bv = ov; bi = oi; }
    }
    if (lane == 0) { s_v[wid] = bv; s_i[wid] = bi; }
    __syncthreads();
    if (wid == 0) {
      float v2 = (lane < 8) ? s_v[lane] : FLT_BIG;
      int   i2 = (lane < 8) ? s_i[lane] : NPTS;
      #pragma unroll
      for (int m = 16; m; m >>= 1) {
        float ov = __shfl_xor(v2, m, 32); int oi = __shfl_xor(i2, m, 32);
        if (ov < v2 || (ov == v2 && oi < i2)) { v2 = ov; i2 = oi; }
      }
      if (lane == 0) s_win = i2;
    }
    __syncthreads();
    int win = s_win;
    if (tid == 0) { knn[g*NK + kk] = b*NPTS + win; d2s[win] = FLT_BIG; }
    __syncthreads();
  }
}

// ---------------------------------------------------------------------------
// 3) Gather features at knn rows -> bf16 A0 (GROWS x 32)
// ---------------------------------------------------------------------------
__global__ __launch_bounds__(256) void gather_kernel(
    const float* __restrict__ features, const int* __restrict__ knn,
    __bf16* __restrict__ A0)
{
  int idx = blockIdx.x * 256 + threadIdx.x;
  if (idx >= GROWS * 32) return;
  int g = idx >> 5, f = idx & 31;
  int row = knn[g];
  A0[idx] = f2bf(features[(size_t)row * 32 + f]);
}

// f32 (K x N) weight -> bf16 transposed (N x K)
__global__ __launch_bounds__(256) void wprep_kernel(
    const float* __restrict__ W, __bf16* __restrict__ Wt, int Kd, int Nd)
{
  int idx = blockIdx.x * 256 + threadIdx.x;
  if (idx >= Kd * Nd) return;
  int k = idx / Nd, n = idx - k * Nd;
  Wt[(size_t)n * Kd + k] = f2bf(W[idx]);
}

// ---------------------------------------------------------------------------
// 4) bf16 WMMA GEMM: out = act(A[MxK] @ Wt^T + bias), Wt stored NxK.
//    One wave -> 16x64 output strip; ping-pong double-buffered fragment loads
//    so VMEM latency overlaps v_wmma issue within a wave.
// ---------------------------------------------------------------------------
__global__ __launch_bounds__(256) void gemm_bf16_wmma(
    const __bf16* __restrict__ A, const __bf16* __restrict__ Wt,
    const float* __restrict__ bias, __bf16* __restrict__ outb,
    float* __restrict__ outf, int M, int N, int Kd, int relu)
{
  const int lane = threadIdx.x & 31;
  const int wave = threadIdx.x >> 5;
  const int tilesN4 = N >> 6;
  const int tiles = (M >> 4) * tilesN4;
  int tile = blockIdx.x * 8 + wave;
  if (tile >= tiles) return;          // wave-uniform: EXEC stays all-ones
  const int tm  = tile / tilesN4;
  const int tn4 = tile % tilesN4;
  const int lm  = lane & 15;
  const int hi  = lane >> 4;          // 0 or 1

  v8f c[4] = {};
  const __bf16* arow  = A  + (size_t)(tm * 16 + lm) * Kd + hi * 8;   // A 16x32 frag base
  const __bf16* bbase = Wt + (size_t)(tn4 * 64 + lm) * Kd + hi * 8;  // B 32x16 frag base

  ABFrag a0, a1, b0[4], b1[4];
  auto load = [&](ABFrag& a, ABFrag* bb, int k0) {
    const __bf16* ap = arow + k0;
    a.q[0] = *(const u32x4*)(ap);
    a.q[1] = *(const u32x4*)(ap + 16);
    #pragma unroll
    for (int j = 0; j < 4; ++j) {
      const __bf16* bp = bbase + (size_t)j * 16 * Kd + k0;
      bb[j].q[0] = *(const u32x4*)(bp);
      bb[j].q[1] = *(const u32x4*)(bp + 16);
    }
  };

  load(a0, b0, 0);
  for (int k0 = 0; k0 < Kd; k0 += 64) {
    if (k0 + 32 < Kd) load(a1, b1, k0 + 32);      // prefetch stage 1
    #pragma unroll
    for (int j = 0; j < 4; ++j)
      c[j] = __builtin_amdgcn_wmma_f32_16x16x32_bf16(
          false, a0.v, false, b0[j].v, (short)0, c[j], false, false);
    if (k0 + 64 < Kd) load(a0, b0, k0 + 64);      // prefetch stage 0
    if (k0 + 32 < Kd) {
      #pragma unroll
      for (int j = 0; j < 4; ++j)
        c[j] = __builtin_amdgcn_wmma_f32_16x16x32_bf16(
            false, a1.v, false, b1[j].v, (short)0, c[j], false, false);
    }
  }

  #pragma unroll
  for (int j = 0; j < 4; ++j) {
    const int n = tn4 * 64 + j * 16 + lm;
    const float bb = bias[n];
    #pragma unroll
    for (int r = 0; r < 8; ++r) {
      float v = c[j][r] + bb;
      if (relu) v = fmaxf(v, 0.0f);
      const int m = tm * 16 + r + hi * 8;          // C/D 16x16 f32 layout
      const size_t off = (size_t)m * N + n;
      if (outf) outf[off] = v;
      else      outb[off] = f2bf(v);
    }
  }
}

// ---------------------------------------------------------------------------
// 5) Max-pool over K=16 gathered rows per token
// ---------------------------------------------------------------------------
__global__ __launch_bounds__(256) void maxpool_kernel(
    const __bf16* __restrict__ PF, __bf16* __restrict__ pooled)
{
  int idx = blockIdx.x * 256 + threadIdx.x;
  if (idx >= NB*NT*TOKD) return;
  int r = idx / TOKD, n = idx - r * TOKD;
  float m = -FLT_BIG; __bf16 mb = f2bf(-FLT_BIG);
  #pragma unroll
  for (int k = 0; k < NK; ++k) {
    __bf16 hb = PF[((size_t)r * NK + k) * TOKD + n];
    float v = bf2f(hb);
    if (v > m) { m = v; mb = hb; }
  }
  pooled[idx] = mb;
}

// ---------------------------------------------------------------------------
extern "C" void kernel_launch(void* const* d_in, const int* in_sizes, int n_in,
                              void* d_out, int out_size, void* d_ws, size_t ws_size,
                              hipStream_t stream) {
  (void)in_sizes; (void)n_in; (void)out_size; (void)ws_size;
  const float* coords   = (const float*)d_in[0];
  const float* features = (const float*)d_in[1];
  const float* times    = (const float*)d_in[3];
  const float* W0 = (const float*)d_in[4];  const float* b0 = (const float*)d_in[5];
  const float* W1 = (const float*)d_in[6];  const float* b1 = (const float*)d_in[7];
  const float* W2 = (const float*)d_in[8];  const float* b2 = (const float*)d_in[9];
  const float* W3 = (const float*)d_in[10]; const float* b3 = (const float*)d_in[11];
  const float* Wn0= (const float*)d_in[12]; const float* bn0= (const float*)d_in[13];
  const float* Wn1= (const float*)d_in[14]; const float* bn1= (const float*)d_in[15];

  float* out_tokens = (float*)d_out;                    // 8*128*768
  float* out_cent   = out_tokens + NB*NT*TOKD;          // 8*128*4
  float* out_mask   = out_cent   + NB*NT*4;             // 8*128

  char* cur = (char*)d_ws;
  auto alloc = [&](size_t bytes)->void* {
    void* r = cur; cur += (bytes + 255) & ~(size_t)255; return r;
  };
  int*    sel    = (int*)   alloc(NB*NT*sizeof(int));
  int*    knn    = (int*)   alloc(GROWS*sizeof(int));
  __bf16* A0     = (__bf16*)alloc((size_t)GROWS*32*2);
  __bf16* H1     = (__bf16*)alloc((size_t)GROWS*256*2);
  __bf16* H2     = (__bf16*)alloc((size_t)GROWS*512*2);
  __bf16* H3     = (__bf16*)alloc((size_t)GROWS*768*2);
  __bf16* PF     = (__bf16*)alloc((size_t)GROWS*768*2);
  __bf16* pooled = (__bf16*)alloc((size_t)NB*NT*TOKD*2);
  __bf16* Ht     = (__bf16*)alloc((size_t)NB*NT*TOKD*2);
  __bf16* Wt0    = (__bf16*)alloc(32*256*2);
  __bf16* Wt1    = (__bf16*)alloc(256*512*2);
  __bf16* Wt2    = (__bf16*)alloc(512*768*2);
  __bf16* Wt3    = (__bf16*)alloc(768*768*2);
  __bf16* Wn0t   = (__bf16*)alloc(768*768*2);
  __bf16* Wn1t   = (__bf16*)alloc(768*768*2);

  // geometry pipeline
  fps_kernel<<<NB, 1024, 0, stream>>>(coords, times, sel, out_cent, out_mask);
  knn_kernel<<<NB*NT, 256, 0, stream>>>(coords, times, out_cent, knn);
  gather_kernel<<<(GROWS*32 + 255)/256, 256, 0, stream>>>(features, knn, A0);

  // weight prep (f32 KxN -> bf16 NxK)
  auto wprep = [&](const float* W, __bf16* Wt, int Kd, int Nd) {
    wprep_kernel<<<(Kd*Nd + 255)/256, 256, 0, stream>>>(W, Wt, Kd, Nd);
  };
  wprep(W0,  Wt0,  32,  256);
  wprep(W1,  Wt1,  256, 512);
  wprep(W2,  Wt2,  512, 768);
  wprep(W3,  Wt3,  768, 768);
  wprep(Wn0, Wn0t, 768, 768);
  wprep(Wn1, Wn1t, 768, 768);

  auto gemm = [&](const __bf16* A, const __bf16* Wt, const float* bias,
                  __bf16* ob, float* of, int M, int N, int K, int relu) {
    int tiles = (M/16) * (N/64);
    gemm_bf16_wmma<<<(tiles + 7)/8, 256, 0, stream>>>(A, Wt, bias, ob, of, M, N, K, relu);
  };
  // per-point MLP on gathered rows only (4x fewer FLOPs than dense pf)
  gemm(A0, Wt0, b0, H1, nullptr, GROWS, 256, 32,  1);
  gemm(H1, Wt1, b1, H2, nullptr, GROWS, 512, 256, 1);
  gemm(H2, Wt2, b2, H3, nullptr, GROWS, 768, 512, 1);
  gemm(H3, Wt3, b3, PF, nullptr, GROWS, 768, 768, 0);

  maxpool_kernel<<<(NB*NT*TOKD + 255)/256, 256, 0, stream>>>(PF, pooled);

  // token MLP
  gemm(pooled, Wn0t, bn0, Ht, nullptr,    NB*NT, 768, 768, 1);
  gemm(Ht,     Wn1t, bn1, nullptr, out_tokens, NB*NT, 768, 768, 0);
}
